// Head_38860864094615
// MI455X (gfx1250) — matrix operand order repaired and checked by reading, not compiled, not run on hardware
//
#include <hip/hip_runtime.h>
#include <hip/hip_bf16.h>

typedef __attribute__((ext_vector_type(16))) __bf16 v16bf;
typedef __attribute__((ext_vector_type(8)))  __bf16 v8bf;
typedef __attribute__((ext_vector_type(8)))  float  v8f;

#define TT 256   // sequence length
#define CC 256   // embedding dim
#define HD 64    // head dim

// ---- LDS layout (bytes). Total 247808 <= 320KB WGP LDS ----
#define OFF_WT   0u        // 3 x transposed weight: 64 rows x 272 bf16 (pad 16 vs bank conflicts)
#define WT_SZ    34816u
#define OFF_RS   104448u   // per-wave scratch: 8 x (16 rows x 68 f32); reused as p-buffer in phase 2
#define RS_WSZ   4352u
#define OFF_Q    139264u   // q: 256 rows x 72 bf16 (pad 8)
#define OFF_K    176128u   // k: 256 rows x 72 bf16
#define OFF_VT   212992u   // v transposed: 64 rows x 272 bf16
#define SMEM_SZ  247808u

#define LOG2E 1.4426950408889634f

__device__ __forceinline__ void lds_fence() {
  asm volatile("s_wait_dscnt 0" ::: "memory");
}

__device__ __forceinline__ float fast_exp(float x) {
  // exp(x) = exp2(x * log2(e)); lowers to v_exp_f32
  return __builtin_amdgcn_exp2f(x * LOG2E);
}

__device__ __forceinline__ v8f zero8() {
  v8f z;
#pragma unroll
  for (int e = 0; e < 8; ++e) z[e] = 0.0f;
  return z;
}

// Build a 16-element bf16 fragment from two aligned 16-byte LDS chunks.
__device__ __forceinline__ v16bf frag16(const unsigned char* p0, const unsigned char* p1) {
  v8bf lo = *(const v8bf*)p0;
  v8bf hi = *(const v8bf*)p1;
  return __builtin_shufflevector(lo, hi, 0,1,2,3,4,5,6,7,8,9,10,11,12,13,14,15);
}

__global__ __launch_bounds__(256, 1)
void fused_rope_attn_kernel(const float* __restrict__ x,
                            const float* __restrict__ Wk,
                            const float* __restrict__ Wq,
                            const float* __restrict__ Wv,
                            float* __restrict__ out) {
  __shared__ __align__(128) unsigned char smem[SMEM_SZ];
  const int b    = blockIdx.x;      // one batch per workgroup
  const int tid  = threadIdx.x;
  const int w    = tid >> 5;        // wave id 0..7 (wave32)
  const int lane = tid & 31;
  const int lr   = lane & 15;       // N / row-in-tile index
  const int hi   = lane >> 4;       // half-wave select

  // ---------------- Stage weights (bf16, transposed) ----------------
  {
    const float* Wm[3] = { Wk, Wq, Wv };
#pragma unroll
    for (int m = 0; m < 3; ++m) {
      const float* W = Wm[m];
      for (int i = 0; i < 64; ++i) {
        int idx = tid + (i << 8);             // coalesced global read
        int kc = idx >> 6, nn = idx & 63;     // W[kc][nn] -> wT[nn][kc]
        *(__bf16*)(smem + OFF_WT + (unsigned)m*WT_SZ + (unsigned)(nn*272 + kc)*2u) = (__bf16)W[idx];
      }
    }
    // zero vT pad columns (cols 256..271) so odd-tile WMMA reads are 0, not NaN
    for (int i = 0; i < 4; ++i) {
      int idx = tid + (i << 8);
      int rr = idx >> 4, ccol = idx & 15;
      *(__bf16*)(smem + OFF_VT + (unsigned)(rr*272 + 256 + ccol)*2u) = (__bf16)0.0f;
    }
  }
  __syncthreads();

  // ---------------- Phase 1: q/k/v projections + RoPE ----------------
  float* rs = (float*)(smem + OFF_RS + (unsigned)w * RS_WSZ);

  for (int tti = 0; tti < 2; ++tti) {
    const int mt = w + tti*8;   // row tile 0..15
    const float* xr = x + ((size_t)b*TT + (size_t)(mt*16 + lr)) * CC;

    // A fragments of x for the whole K=256, loaded once, bf16 in registers.
    v16bf a[8];
#pragma unroll
    for (int kk = 0; kk < 8; ++kk) {
      const int kb = kk*32 + hi*8;            // A-layout: lane<16 -> K kb..kb+7 & kb+16..kb+23
      float4 f0 = *(const float4*)(xr + kb);
      float4 f1 = *(const float4*)(xr + kb + 4);
      float4 f2 = *(const float4*)(xr + kb + 16);
      float4 f3 = *(const float4*)(xr + kb + 20);
      v16bf t;
      t[0]=(__bf16)f0.x;  t[1]=(__bf16)f0.y;  t[2]=(__bf16)f0.z;  t[3]=(__bf16)f0.w;
      t[4]=(__bf16)f1.x;  t[5]=(__bf16)f1.y;  t[6]=(__bf16)f1.z;  t[7]=(__bf16)f1.w;
      t[8]=(__bf16)f2.x;  t[9]=(__bf16)f2.y;  t[10]=(__bf16)f2.z; t[11]=(__bf16)f2.w;
      t[12]=(__bf16)f3.x; t[13]=(__bf16)f3.y; t[14]=(__bf16)f3.z; t[15]=(__bf16)f3.w;
      a[kk] = t;
    }

#pragma unroll
    for (int m = 0; m < 3; ++m) {             // 0=K, 1=Q, 2=V
      v8f acc[4];
#pragma unroll
      for (int nt = 0; nt < 4; ++nt) acc[nt] = zero8();
#pragma unroll
      for (int kk = 0; kk < 8; ++kk) {
#pragma unroll
        for (int nt = 0; nt < 4; ++nt) {
          const unsigned boff = OFF_WT + (unsigned)m*WT_SZ
                              + (unsigned)((nt*16 + lr)*272 + kk*32 + hi*8)*2u;
          v16bf bf = frag16(smem + boff, smem + boff + 32u);
          acc[nt] = __builtin_amdgcn_wmma_f32_16x16x32_bf16(
                      false, a[kk], false, bf, (short)0, acc[nt], false, false);
        }
      }
      if (m == 2) {
        // V: convert to bf16 and store transposed (vT[col][row]) — one b128 store per nt
#pragma unroll
        for (int nt = 0; nt < 4; ++nt) {
          v8bf pk;
#pragma unroll
          for (int r = 0; r < 8; ++r) pk[r] = (__bf16)acc[nt][r];
          *(v8bf*)(smem + OFF_VT + (unsigned)((nt*16 + lr)*272 + mt*16 + hi*8)*2u) = pk;
        }
      } else {
        // K/Q: raw f32 to per-wave scratch (D layout: row = r+8*hi, col = nt*16+lr)
#pragma unroll
        for (int nt = 0; nt < 4; ++nt)
#pragma unroll
          for (int r = 0; r < 8; ++r)
            rs[(r + 8*hi)*68 + nt*16 + lr] = acc[nt][r];
        lds_fence();
        // RoPE (row-local mixing of column pairs) -> bf16 row-major dest
        unsigned char* dst = smem + ((m == 0) ? OFF_K : OFF_Q);
        const int row = lr;
        const int t   = mt*16 + lr;           // sequence position
        const float tf = (float)t;
#pragma unroll
        for (int dd = 0; dd < 16; ++dd) {
          const int d = hi*16 + dd;           // d in 0..31
          float x1 = rs[row*68 + 2*d];
          float x2 = rs[row*68 + 2*d + 1];
          // freq = 10000^(-d/32) = exp2(-d * log2(1e4)/32)
          float ang = tf * __builtin_amdgcn_exp2f((float)d * -0.4152410119f);
          float sv, cv;
          __sincosf(ang, &sv, &cv);
          *(__bf16*)(dst + (unsigned)(t*72 + d)*2u)      = (__bf16)(x1*cv - x2*sv);
          *(__bf16*)(dst + (unsigned)(t*72 + 32 + d)*2u) = (__bf16)(x1*sv + x2*cv);
        }
        lds_fence();
      }
    }
  }
  __syncthreads();

  // ---------------- Phase 2: causal flash attention ----------------
  __bf16* pb = (__bf16*)(smem + OFF_RS + (unsigned)w * RS_WSZ);  // 16 x 40 bf16 p-buffer
  const float SCALE = 0.0625f;   // C^-0.5 = 1/16

  for (int qi = 0; qi < 2; ++qi) {
    const int mt = qi ? (15 - w) : w;   // pair tiles for load balance (17 key tiles each)

    // q A-fragments (K = Hd = 64 -> two K=32 fragments)
    v16bf aq[2];
#pragma unroll
    for (int kk = 0; kk < 2; ++kk) {
      const unsigned qoff = OFF_Q + (unsigned)((mt*16 + lr)*72 + kk*32 + hi*8)*2u;
      aq[kk] = frag16(smem + qoff, smem + qoff + 32u);
    }

    v8f oacc[4];
#pragma unroll
    for (int nt = 0; nt < 4; ++nt) oacc[nt] = zero8();
    float mst[8], lst[8];
#pragma unroll
    for (int r = 0; r < 8; ++r) { mst[r] = -1e30f; lst[r] = 0.0f; }

    for (int j = 0; j <= mt; j += 2) {          // 32 keys per step
      const bool has2 = (j + 1) <= mt;          // wave-uniform branch: EXEC stays full

      // scores s = q @ k^T  (B fragments contiguous from row-major k_lds)
      v8f s0 = zero8();
#pragma unroll
      for (int kk = 0; kk < 2; ++kk) {
        const unsigned koff = OFF_K + (unsigned)((j*16 + lr)*72 + kk*32 + hi*8)*2u;
        v16bf bk = frag16(smem + koff, smem + koff + 32u);
        s0 = __builtin_amdgcn_wmma_f32_16x16x32_bf16(
               false, aq[kk], false, bk, (short)0, s0, false, false);
      }
      v8f s1 = zero8();
      if (has2) {
#pragma unroll
        for (int kk = 0; kk < 2; ++kk) {
          const unsigned koff = OFF_K + (unsigned)(((j+1)*16 + lr)*72 + kk*32 + hi*8)*2u;
          v16bf bk = frag16(smem + koff, smem + koff + 32u);
          s1 = __builtin_amdgcn_wmma_f32_16x16x32_bf16(
                 false, aq[kk], false, bk, (short)0, s1, false, false);
        }
      }

      // online softmax in f32 (per row r+8*hi; reductions across the 16-lane half)
      float p0v[8], p1v[8], alp[8];
#pragma unroll
      for (int r = 0; r < 8; ++r) {
        const int rowg = mt*16 + r + 8*hi;
        float a0 = (j*16 + lr <= rowg)                 ? s0[r]*SCALE : -1e30f;
        float a1 = (has2 && ((j+1)*16 + lr <= rowg))   ? s1[r]*SCALE : -1e30f;
        float mx = fmaxf(a0, a1);
        mx = fmaxf(mx, __shfl_xor(mx, 1));
        mx = fmaxf(mx, __shfl_xor(mx, 2));
        mx = fmaxf(mx, __shfl_xor(mx, 4));
        mx = fmaxf(mx, __shfl_xor(mx, 8));
        const float mnew = fmaxf(mst[r], mx);
        const float al = fast_exp(mst[r] - mnew);
        const float e0 = fast_exp(a0 - mnew);
        const float e1 = fast_exp(a1 - mnew);
        float ps = e0 + e1;
        ps += __shfl_xor(ps, 1);
        ps += __shfl_xor(ps, 2);
        ps += __shfl_xor(ps, 4);
        ps += __shfl_xor(ps, 8);
        lst[r] = lst[r]*al + ps;
        mst[r] = mnew;
        alp[r] = al; p0v[r] = e0; p1v[r] = e1;
      }
#pragma unroll
      for (int nt = 0; nt < 4; ++nt)
#pragma unroll
        for (int r = 0; r < 8; ++r) oacc[nt][r] *= alp[r];

      // D-layout -> A-layout bounce of probabilities through per-wave LDS (bf16)
#pragma unroll
      for (int r = 0; r < 8; ++r) {
        const int prow = r + 8*hi;
        pb[prow*40 + lr]      = (__bf16)p0v[r];
        pb[prow*40 + 16 + lr] = (__bf16)p1v[r];
      }
      lds_fence();

      // out += P(16x32) @ V(32x64): B fragments contiguous from vT
      const unsigned poff = (unsigned)(lr*40 + hi*8)*2u;
      v16bf ap = frag16((const unsigned char*)pb + poff, (const unsigned char*)pb + poff + 32u);
#pragma unroll
      for (int nt = 0; nt < 4; ++nt) {
        const unsigned voff = OFF_VT + (unsigned)((nt*16 + lr)*272 + j*16 + hi*8)*2u;
        v16bf bv = frag16(smem + voff, smem + voff + 32u);
        oacc[nt] = __builtin_amdgcn_wmma_f32_16x16x32_bf16(
                     false, ap, false, bv, (short)0, oacc[nt], false, false);
      }
    }

    // normalize and write f32 output (lanes 0..15 store 64B-contiguous chunks)
#pragma unroll
    for (int r = 0; r < 8; ++r) {
      const float inv = 1.0f / lst[r];
      const int rowg = mt*16 + r + 8*hi;
      float* op = out + ((size_t)b*TT + rowg)*HD;
#pragma unroll
      for (int nt = 0; nt < 4; ++nt)
        op[nt*16 + lr] = oacc[nt][r] * inv;
    }
  }
}

extern "C" void kernel_launch(void* const* d_in, const int* in_sizes, int n_in,
                              void* d_out, int out_size, void* d_ws, size_t ws_size,
                              hipStream_t stream) {
  const float* x  = (const float*)d_in[0];
  const float* Wk = (const float*)d_in[1];
  const float* Wq = (const float*)d_in[2];
  const float* Wv = (const float*)d_in[3];
  float* out = (float*)d_out;
  (void)in_sizes; (void)n_in; (void)out_size; (void)d_ws; (void)ws_size;
  dim3 grid(512), block(256);   // one batch per workgroup, 8 wave32 per WGP
  fused_rope_attn_kernel<<<grid, block, 0, stream>>>(x, Wk, Wq, Wv, out);
}
